// SeriesLSTM_42193758716329
// MI455X (gfx1250) — matrix operand order repaired and checked by reading, not compiled, not run on hardware
//
#include <hip/hip_runtime.h>
#include <hip/hip_bf16.h>

// Problem dims (fixed by reference)
#define BDIM 128
#define TDIM 256
#define IDIM 64
#define HDIM 512
#define GDIM 2048   // 4*H

typedef __attribute__((ext_vector_type(16))) __bf16 v16bf;
typedef __attribute__((ext_vector_type(8)))  __bf16 v8bf;
typedef __attribute__((ext_vector_type(8)))  float  v8f;

__device__ __forceinline__ float sigmoidf_(float x) {
  return 1.0f / (1.0f + __expf(-x));
}

// A-operand fragment (16x32 bf16, wave32 layout):
//   lanes 0-15 / 16-31: M = lane&15, K-halves selected by lane>>4.
//   element i<8  -> k = kbase + half*8 + i        (VGPR0-3)
//   element i>=8 -> k = kbase + 16 + half*8 + i-8 (VGPR4-7)
// Row-major bf16 activations satisfy this with two 16-byte loads per lane.
__device__ __forceinline__ v16bf load_a_frag(const __bf16* base, int row,
                                             int row_stride, int kbase, int half) {
  const __bf16* p0 = base + (size_t)row * row_stride + kbase + half * 8;
  v8bf lo = *reinterpret_cast<const v8bf*>(p0);
  v8bf hi = *reinterpret_cast<const v8bf*>(p0 + 16);
  v16bf a;
#pragma unroll
  for (int i = 0; i < 8; ++i) { a[i] = lo[i]; a[8 + i] = hi[i]; }
  return a;
}

// B-operand fragment (32x16 bf16): pre-packed so each lane reads its 16
// halves (32 bytes) contiguously: offset = ((nt*KT + kt)*32 + lane)*16.
__device__ __forceinline__ v16bf load_b_frag(const __bf16* wp, int nt, int kt,
                                             int KT, int lane) {
  const __bf16* p = wp + ((((size_t)nt * KT + kt) * 32 + lane) << 4);
  v8bf lo = *reinterpret_cast<const v8bf*>(p);
  v8bf hi = *reinterpret_cast<const v8bf*>(p + 8);
  v16bf b;
#pragma unroll
  for (int i = 0; i < 8; ++i) { b[i] = lo[i]; b[8 + i] = hi[i]; }
  return b;
}

// One bidirectional LSTM timestep.
// gates[B,4H] = A1 @ W1^T + h_prev @ W2^T + bias; then cell update.
// Each wave: one (mt, jt) 16x16 tile, all 4 gate columns (i,f,g,o).
// Grid: 64 blocks x 256 threads = 512 waves = 2 dirs x 8 mt x 32 jt.
__global__ void __launch_bounds__(256)
lstm_step_kernel(const __bf16* __restrict__ a1_f, const __bf16* __restrict__ a1_b,
                 int KT1, int a1_stride,
                 const __bf16* __restrict__ hprev, __bf16* __restrict__ hnext,
                 float* __restrict__ cstate,
                 const __bf16* __restrict__ w1_f, const __bf16* __restrict__ w1_b,
                 const __bf16* __restrict__ w2_f, const __bf16* __restrict__ w2_b,
                 const float* __restrict__ bias_f, const float* __restrict__ bias_b,
                 __bf16* __restrict__ seq_f, __bf16* __restrict__ seq_b,
                 int seq_stride) {
  const int lane = threadIdx.x & 31;
  const int wg   = blockIdx.x * 8 + (threadIdx.x >> 5);
  const int dir  = wg >> 8;        // 0 = forward, 1 = backward
  const int rem  = wg & 255;
  const int mt   = rem >> 5;       // 0..7  (batch tile)
  const int jt   = rem & 31;       // 0..31 (hidden-column tile)

  const __bf16* a1   = dir ? a1_b   : a1_f;
  const __bf16* w1   = dir ? w1_b   : w1_f;
  const __bf16* w2   = dir ? w2_b   : w2_f;
  const float*  bias = dir ? bias_b : bias_f;
  __bf16*       seq  = dir ? seq_b  : seq_f;
  const __bf16* hp   = hprev  + (size_t)dir * (BDIM * HDIM);
  __bf16*       hn   = hnext  + (size_t)dir * (BDIM * HDIM);
  float*        cs   = cstate + (size_t)dir * (BDIM * HDIM);

  const int half = lane >> 4;
  const int arow = mt * 16 + (lane & 15);

  v8f acc0 = {}, acc1 = {}, acc2 = {}, acc3 = {};

  // Input contribution: K = 32*KT1
  for (int kt = 0; kt < KT1; ++kt) {
    v16bf a = load_a_frag(a1, arow, a1_stride, kt * 32, half);
    if (kt + 1 < KT1)  // stream-hint next weight tile (L2-resident)
      __builtin_prefetch(w1 + ((((size_t)jt * KT1 + (kt + 1)) * 32 + lane) << 4), 0, 1);
    v16bf b0 = load_b_frag(w1, 0 * 32 + jt, kt, KT1, lane);
    v16bf b1 = load_b_frag(w1, 1 * 32 + jt, kt, KT1, lane);
    v16bf b2 = load_b_frag(w1, 2 * 32 + jt, kt, KT1, lane);
    v16bf b3 = load_b_frag(w1, 3 * 32 + jt, kt, KT1, lane);
    acc0 = __builtin_amdgcn_wmma_f32_16x16x32_bf16(false, a, false, b0, (short)0, acc0, false, false);
    acc1 = __builtin_amdgcn_wmma_f32_16x16x32_bf16(false, a, false, b1, (short)0, acc1, false, false);
    acc2 = __builtin_amdgcn_wmma_f32_16x16x32_bf16(false, a, false, b2, (short)0, acc2, false, false);
    acc3 = __builtin_amdgcn_wmma_f32_16x16x32_bf16(false, a, false, b3, (short)0, acc3, false, false);
  }

  // Recurrent contribution: K = H = 512 -> 16 k-tiles
  for (int kt = 0; kt < 16; ++kt) {
    v16bf a = load_a_frag(hp, arow, HDIM, kt * 32, half);
    v16bf b0 = load_b_frag(w2, 0 * 32 + jt, kt, 16, lane);
    v16bf b1 = load_b_frag(w2, 1 * 32 + jt, kt, 16, lane);
    v16bf b2 = load_b_frag(w2, 2 * 32 + jt, kt, 16, lane);
    v16bf b3 = load_b_frag(w2, 3 * 32 + jt, kt, 16, lane);
    acc0 = __builtin_amdgcn_wmma_f32_16x16x32_bf16(false, a, false, b0, (short)0, acc0, false, false);
    acc1 = __builtin_amdgcn_wmma_f32_16x16x32_bf16(false, a, false, b1, (short)0, acc1, false, false);
    acc2 = __builtin_amdgcn_wmma_f32_16x16x32_bf16(false, a, false, b2, (short)0, acc2, false, false);
    acc3 = __builtin_amdgcn_wmma_f32_16x16x32_bf16(false, a, false, b3, (short)0, acc3, false, false);
  }

  // Epilogue: C/D layout -> lane holds column n, rows mbase..mbase+7
  const int n = jt * 16 + (lane & 15);
  const float bi = bias[n];
  const float bf = bias[HDIM + n];
  const float bg = bias[2 * HDIM + n];
  const float bo = bias[3 * HDIM + n];
  const int mbase = mt * 16 + half * 8;
#pragma unroll
  for (int g = 0; g < 8; ++g) {
    const int m = mbase + g;
    const float iv = sigmoidf_(acc0[g] + bi);
    const float fv = sigmoidf_(acc1[g] + bf);
    const float gv = tanhf(acc2[g] + bg);
    const float ov = sigmoidf_(acc3[g] + bo);
    const float cp = cs[m * HDIM + n];
    const float cn = fv * cp + iv * gv;
    const float hv = ov * tanhf(cn);
    cs[m * HDIM + n] = cn;
    const __bf16 hb = (__bf16)hv;
    hn[m * HDIM + n] = hb;
    if (seq) seq[(size_t)m * seq_stride + n] = hb;
  }
}

// fp32 [G,K] row-major -> bf16 packed B-operand tiles (wave32 layout).
// lane L: n = L&15, k = kt*32 + (L>>4)*8 + (elem&7) + (elem>>3)*16 (elem = 0..15).
__global__ void pack_w_kernel(const float* __restrict__ W, __bf16* __restrict__ out,
                              int K, int KT) {
  const int idx = blockIdx.x * blockDim.x + threadIdx.x;
  if (idx >= GDIM * K) return;
  const int elem = idx & 15;
  const int lane = (idx >> 4) & 31;
  const int tile = idx >> 9;
  const int kt = tile % KT;
  const int nt = tile / KT;
  const int nn = lane & 15;
  const int hf = lane >> 4;
  const int k  = kt * 32 + hf * 8 + (elem & 7) + ((elem >> 3) << 4);
  out[idx] = (__bf16)W[(size_t)(nt * 16 + nn) * K + k];
}

__global__ void cvt_bf16_kernel(const float* __restrict__ in, __bf16* __restrict__ out, int n) {
  const int i = blockIdx.x * blockDim.x + threadIdx.x;
  if (i < n) out[i] = (__bf16)in[i];
}

__global__ void zero_kernel(uint4* __restrict__ p, int n16) {
  const int i = blockIdx.x * blockDim.x + threadIdx.x;
  if (i < n16) p[i] = uint4{0u, 0u, 0u, 0u};
}

// out[b] = dot(concat(h1f_last[b], h1b_last[b]), fc_w) + fc_b
__global__ void fc_kernel(const __bf16* __restrict__ h1last, const float* __restrict__ fcw,
                          const float* __restrict__ fcb, float* __restrict__ out) {
  __shared__ float red[256];
  const int b = blockIdx.x, tid = threadIdx.x;
  float s = 0.0f;
  for (int j = tid; j < HDIM; j += 256) {
    s += (float)h1last[(size_t)b * HDIM + j] * fcw[j];
    s += (float)h1last[(size_t)BDIM * HDIM + (size_t)b * HDIM + j] * fcw[HDIM + j];
  }
  red[tid] = s;
  __syncthreads();
  for (int off = 128; off > 0; off >>= 1) {
    if (tid < off) red[tid] += red[tid + off];
    __syncthreads();
  }
  if (tid == 0) out[b] = red[0] + fcb[0];
}

extern "C" void kernel_launch(void* const* d_in, const int* in_sizes, int n_in,
                              void* d_out, int out_size, void* d_ws, size_t ws_size,
                              hipStream_t stream) {
  (void)in_sizes; (void)n_in; (void)out_size; (void)ws_size;

  const float* x       = (const float*)d_in[0];
  const float* w_ih_0f = (const float*)d_in[1];
  const float* w_hh_0f = (const float*)d_in[2];
  const float* b_0f    = (const float*)d_in[3];
  const float* w_ih_0b = (const float*)d_in[4];
  const float* w_hh_0b = (const float*)d_in[5];
  const float* b_0b    = (const float*)d_in[6];
  const float* w_ih_1f = (const float*)d_in[7];
  const float* w_hh_1f = (const float*)d_in[8];
  const float* b_1f    = (const float*)d_in[9];
  const float* w_ih_1b = (const float*)d_in[10];
  const float* w_hh_1b = (const float*)d_in[11];
  const float* b_1b    = (const float*)d_in[12];
  const float* fc_w    = (const float*)d_in[13];
  const float* fc_b    = (const float*)d_in[14];
  float* out = (float*)d_out;

  // Workspace carve-up (~90 MB total)
  char* base = (char*)d_ws;
  size_t off = 0;
  auto alloc = [&](size_t bytes) -> char* {
    off = (off + 255) & ~(size_t)255;
    char* p = base + off;
    off += bytes;
    return p;
  };

  __bf16* x_bf    = (__bf16*)alloc((size_t)BDIM * TDIM * IDIM * 2);      // 4.2 MB
  __bf16* pw_ih0f = (__bf16*)alloc((size_t)GDIM * IDIM * 2);
  __bf16* pw_hh0f = (__bf16*)alloc((size_t)GDIM * HDIM * 2);
  __bf16* pw_ih0b = (__bf16*)alloc((size_t)GDIM * IDIM * 2);
  __bf16* pw_hh0b = (__bf16*)alloc((size_t)GDIM * HDIM * 2);
  __bf16* pw_ih1f = (__bf16*)alloc((size_t)GDIM * 2 * HDIM * 2);
  __bf16* pw_hh1f = (__bf16*)alloc((size_t)GDIM * HDIM * 2);
  __bf16* pw_ih1b = (__bf16*)alloc((size_t)GDIM * 2 * HDIM * 2);
  __bf16* pw_hh1b = (__bf16*)alloc((size_t)GDIM * HDIM * 2);
  __bf16* h0seq   = (__bf16*)alloc((size_t)BDIM * TDIM * 2 * HDIM * 2);  // 67 MB
  __bf16* hbuf    = (__bf16*)alloc((size_t)2 * 2 * BDIM * HDIM * 2);     // pingpong x dir
  float*  cbuf    = (float*)alloc((size_t)2 * BDIM * HDIM * 4);          // dir
  __bf16* h1last  = (__bf16*)alloc((size_t)2 * BDIM * HDIM * 2);         // dir

  const int PP = 2 * BDIM * HDIM;  // elements per pingpong slab (both dirs)

  // ---- pack inputs & weights (bf16) ----
  {
    const int n = BDIM * TDIM * IDIM;
    cvt_bf16_kernel<<<(n + 255) / 256, 256, 0, stream>>>(x, x_bf, n);
  }
  pack_w_kernel<<<(GDIM * IDIM      + 255) / 256, 256, 0, stream>>>(w_ih_0f, pw_ih0f, IDIM,     IDIM / 32);
  pack_w_kernel<<<(GDIM * HDIM      + 255) / 256, 256, 0, stream>>>(w_hh_0f, pw_hh0f, HDIM,     HDIM / 32);
  pack_w_kernel<<<(GDIM * IDIM      + 255) / 256, 256, 0, stream>>>(w_ih_0b, pw_ih0b, IDIM,     IDIM / 32);
  pack_w_kernel<<<(GDIM * HDIM      + 255) / 256, 256, 0, stream>>>(w_hh_0b, pw_hh0b, HDIM,     HDIM / 32);
  pack_w_kernel<<<(GDIM * 2 * HDIM  + 255) / 256, 256, 0, stream>>>(w_ih_1f, pw_ih1f, 2 * HDIM, 2 * HDIM / 32);
  pack_w_kernel<<<(GDIM * HDIM      + 255) / 256, 256, 0, stream>>>(w_hh_1f, pw_hh1f, HDIM,     HDIM / 32);
  pack_w_kernel<<<(GDIM * 2 * HDIM  + 255) / 256, 256, 0, stream>>>(w_ih_1b, pw_ih1b, 2 * HDIM, 2 * HDIM / 32);
  pack_w_kernel<<<(GDIM * HDIM      + 255) / 256, 256, 0, stream>>>(w_hh_1b, pw_hh1b, HDIM,     HDIM / 32);

  const int hz = (2 * PP * 2) / 16;           // hbuf bytes / 16
  const int cz = (2 * BDIM * HDIM * 4) / 16;  // cbuf bytes / 16

  // ---- Layer 0 (bidirectional, fused dirs, 256 sequential steps) ----
  zero_kernel<<<(hz + 255) / 256, 256, 0, stream>>>((uint4*)hbuf, hz);
  zero_kernel<<<(cz + 255) / 256, 256, 0, stream>>>((uint4*)cbuf, cz);
  for (int s = 0; s < TDIM; ++s) {
    const int tf = s, tb = TDIM - 1 - s;
    lstm_step_kernel<<<64, 256, 0, stream>>>(
        x_bf + (size_t)tf * IDIM, x_bf + (size_t)tb * IDIM,
        IDIM / 32, TDIM * IDIM,
        hbuf + (size_t)(s & 1) * PP, hbuf + (size_t)((s + 1) & 1) * PP, cbuf,
        pw_ih0f, pw_ih0b, pw_hh0f, pw_hh0b, b_0f, b_0b,
        h0seq + (size_t)tf * (2 * HDIM),
        h0seq + (size_t)tb * (2 * HDIM) + HDIM,
        TDIM * 2 * HDIM);
  }

  // ---- Layer 1 (bidirectional; only h[:, T-1, :] is consumed downstream) ----
  zero_kernel<<<(hz + 255) / 256, 256, 0, stream>>>((uint4*)hbuf, hz);
  zero_kernel<<<(cz + 255) / 256, 256, 0, stream>>>((uint4*)cbuf, cz);
  for (int s = 0; s < TDIM; ++s) {
    const int tf = s, tb = TDIM - 1 - s;
    lstm_step_kernel<<<64, 256, 0, stream>>>(
        h0seq + (size_t)tf * (2 * HDIM), h0seq + (size_t)tb * (2 * HDIM),
        2 * HDIM / 32, TDIM * 2 * HDIM,
        hbuf + (size_t)(s & 1) * PP, hbuf + (size_t)((s + 1) & 1) * PP, cbuf,
        pw_ih1f, pw_ih1b, pw_hh1f, pw_hh1b, b_1f, b_1b,
        h1last,                                        // fwd: overwrite -> ends at t=T-1
        (s == 0) ? (h1last + (size_t)BDIM * HDIM) : (__bf16*)nullptr,  // bwd t=T-1
        HDIM);
  }

  // ---- FC head ----
  fc_kernel<<<BDIM, 256, 0, stream>>>(h1last, fc_w, fc_b, out);
}
